// RNNPool_33157147525755
// MI455X (gfx1250) — compile-verified, matrix operand order checked by reading, not run on hardware
//
#include <hip/hip_runtime.h>
#include <hip/hip_bf16.h>

typedef __attribute__((ext_vector_type(16))) __bf16        v16bf;
typedef __attribute__((ext_vector_type(8)))  float         v8f;
typedef __attribute__((ext_vector_type(4)))  unsigned int  v4u;
typedef __attribute__((ext_vector_type(4)))  float         v4f;

#define EDIM   512
#define KSTEPS 8
#define KTOT   1024           // fused K = [x_t | h]
#define APAD   8              // u16 pad per row -> +16B, breaks LDS bank conflicts
#define WPACK_U16 (32*32*512) // 32 n-tiles * 32 k-chunks * (32 lanes * 16 elems)
#define TWO_LOG2E 2.8853900817779268f   // 2*log2(e): tanh(z) = 1 - 2/(exp2(z*TWO_LOG2E)+1)

__device__ __forceinline__ unsigned short f32_to_bf16(float f) {
  unsigned int u = __float_as_uint(f);
  u += 0x7FFFu + ((u >> 16) & 1u);          // round-to-nearest-even
  return (unsigned short)(u >> 16);
}

// ---------------------------------------------------------------------------
// Prep: pack [W_ih ; W_hh] (f32, [n=512, k]) -> bf16 tiles in exact WMMA-B
// lane/register order, so the main kernel fetches B with coalesced b128 loads.
// Tile (nt, kc): 32x16 (KxN) slice, 512 u16. Element order inside a tile:
//   u16 index = q*256 + lane*8 + j,  e = q*8 + j (position in v16bf),
//   n = nt*16 + (lane&15),  k = kc*32 + (lane>>4)*8 + e + (e>=8 ? 8 : 0)
// Also fuses bias = b_ih + b_hh (f32) after the 1MB weight block.
// ---------------------------------------------------------------------------
__global__ void rnnpool_pack(const float* __restrict__ W_ih,
                             const float* __restrict__ W_hh,
                             const float* __restrict__ b_ih,
                             const float* __restrict__ b_hh,
                             unsigned short* __restrict__ Wp,
                             float* __restrict__ biasc) {
  int tid = blockIdx.x * blockDim.x + threadIdx.x;
  if (tid < WPACK_U16) {
    int tix  = tid >> 9;          // (nt*32 + kc)
    int w    = tid & 511;
    int q    = w >> 8;
    int lane = (w >> 3) & 31;
    int j    = w & 7;
    int e    = q * 8 + j;
    int g    = lane >> 4;
    int k_local = g * 8 + e + (e >= 8 ? 8 : 0);
    int nt = tix >> 5;
    int kc = tix & 31;
    int n  = nt * 16 + (lane & 15);
    int k  = kc * 32 + k_local;
    float v = (k < EDIM) ? W_ih[n * EDIM + k] : W_hh[n * EDIM + (k - EDIM)];
    Wp[tid] = f32_to_bf16(v);
  } else if (tid < WPACK_U16 + EDIM) {
    int i = tid - WPACK_U16;
    biasc[i] = b_ih[i] + b_hh[i];
  }
}

// ---------------------------------------------------------------------------
// Main: one WG = 32 windows, 512 threads = 16 wave32s.
// Wave (wm, wn): rows wm*16..+16, cols wn*64..+64 (4 WMMA N-tiles).
// Per step: x_t -> LDS A-panel (bf16, NT loads), K-loop of
// v_wmma_f32_16x16x32_bf16 over fused K (512 on step 0, 1024 after),
// branchless tanh epilogue (v_fma / v_exp / v_rcp) writes h back to the
// panel (bf16) or to d_out (f32, NT stores) on the last step.
// ---------------------------------------------------------------------------
__global__ __launch_bounds__(512) void rnnpool_main(
    const float* __restrict__ x,
    const unsigned short* __restrict__ Wp_u16,
    const float* __restrict__ biasc,
    float* __restrict__ out) {
  __shared__ unsigned short a_panel[32][KTOT + APAD];

  const int tid  = threadIdx.x;
  const int lane = tid & 31;
  const int wid  = tid >> 5;
  const int wm   = wid & 1;        // row tile 0..1
  const int wn   = wid >> 1;       // col group 0..7 (64 cols each)
  const int g    = lane >> 4;
  const int ln16 = lane & 15;

  const v4u* Wp = (const v4u*)Wp_u16;

  // bias pre-scaled by 2*log2(e): exp2 argument = fma(acc, TWO_LOG2E, biasL)
  float biasL[4];
  #pragma unroll
  for (int nt = 0; nt < 4; ++nt)
    biasL[nt] = biasc[wn * 64 + nt * 16 + ln16] * TWO_LOG2E;

  // cooperative x-load coordinates: 32 rows x 512 f32 per step
  const int row_l_x = tid >> 4;            // 0..31
  const int e0      = (tid & 15) * 32;     // 32 f32 per thread
  const int row_g_x = blockIdx.x * 32 + row_l_x;    // window index bd*512 + w
  const long xrow   = (long)(row_g_x >> 9) * 4096 + (long)(row_g_x & 511) * 8;

  const int arow = wm * 16 + ln16;         // A-matrix row this lane feeds

  for (int t = 0; t < KSTEPS; ++t) {
    // ---- load x_t slab (non-temporal: streamed once) into a_panel[:,0:512] ----
    {
      const v4f* xp = (const v4f*)(x + (xrow + t) * EDIM + e0);
      #pragma unroll
      for (int j = 0; j < 8; ++j) {
        v4f v = __builtin_nontemporal_load(xp + j);
        unsigned int p0 = (unsigned int)f32_to_bf16(v.x) |
                          ((unsigned int)f32_to_bf16(v.y) << 16);
        unsigned int p1 = (unsigned int)f32_to_bf16(v.z) |
                          ((unsigned int)f32_to_bf16(v.w) << 16);
        *(uint2*)&a_panel[row_l_x][e0 + j * 4] = make_uint2(p0, p1);
      }
    }
    __syncthreads();

    // ---- GEMM: acc[m,n] = sum_k A[m,k] * Wcomb[n,k] ----
    v8f acc[4] = {};
    const int kc_end = (t == 0) ? 16 : 32;   // step 0: h == 0, skip its K half
    for (int kc = 0; kc < kc_end; ++kc) {
      union { v16bf v; v4u q[2]; } A;
      const unsigned short* ap = &a_panel[arow][kc * 32 + g * 8];
      A.q[0] = *(const v4u*)ap;          // K = kbase + g*8 .. +7
      A.q[1] = *(const v4u*)(ap + 16);   // K = kbase + 16 + g*8 .. +7
      #pragma unroll
      for (int nt = 0; nt < 4; ++nt) {
        union { v16bf v; v4u q[2]; } Bm;
        const int tix = (wn * 4 + nt) * 32 + kc;
        Bm.q[0] = Wp[tix * 64 + lane];
        Bm.q[1] = Wp[tix * 64 + 32 + lane];
        acc[nt] = __builtin_amdgcn_wmma_f32_16x16x32_bf16(
            false, A.v, false, Bm.v, (short)0, acc[nt], false, false);
      }
    }
    __syncthreads();

    // ---- epilogue: h = tanh(acc + bias) = 1 - 2*rcp(exp2(fma(...))+1) ----
    const bool last = (t == KSTEPS - 1);
    #pragma unroll
    for (int nt = 0; nt < 4; ++nt) {
      const int col = wn * 64 + nt * 16 + ln16;
      #pragma unroll
      for (int r = 0; r < 8; ++r) {
        const int row = wm * 16 + r + 8 * g;      // C/D layout: VGPR r, lane grp
        const float arg = __builtin_fmaf(acc[nt][r], TWO_LOG2E, biasL[nt]);
        const float e2  = __builtin_amdgcn_exp2f(arg);       // v_exp_f32
        const float hval = 1.0f - 2.0f * __builtin_amdgcn_rcpf(e2 + 1.0f);
        if (last) {
          const int row_g = blockIdx.x * 32 + row;
          __builtin_nontemporal_store(hval, &out[(long)row_g * EDIM + col]);
        } else {
          a_panel[row][EDIM + col] = f32_to_bf16(hval);
        }
      }
    }
    __syncthreads();
  }
}

extern "C" void kernel_launch(void* const* d_in, const int* in_sizes, int n_in,
                              void* d_out, int out_size, void* d_ws, size_t ws_size,
                              hipStream_t stream) {
  const float* x    = (const float*)d_in[0];
  const float* W_ih = (const float*)d_in[1];
  const float* W_hh = (const float*)d_in[2];
  const float* b_ih = (const float*)d_in[3];
  const float* b_hh = (const float*)d_in[4];

  unsigned short* Wp = (unsigned short*)d_ws;                       // 1 MB bf16
  float* biasc = (float*)((char*)d_ws + WPACK_U16 * sizeof(unsigned short));

  const int prep_threads = WPACK_U16 + EDIM;
  rnnpool_pack<<<(prep_threads + 255) / 256, 256, 0, stream>>>(
      W_ih, W_hh, b_ih, b_hh, Wp, biasc);

  // 8192 windows / 32 per WG = 256 workgroups of 512 threads (16 wave32s)
  rnnpool_main<<<256, 512, 0, stream>>>(x, Wp, biasc, (float*)d_out);
}
